// NODE_32933809225749
// MI455X (gfx1250) — matrix operand order repaired and checked by reading, not compiled, not run on hardware
//
#include <hip/hip_runtime.h>
#include <hip/hip_bf16.h>

// ---------------------------------------------------------------------------
// GRU-ODE recurrent net on gfx1250.
// Matmuls: v_wmma_f32_16x16x32_f16 (f16 in, f32 accumulate).
// One persistent kernel: 128 blocks x 16 rows, 8 waves/block, T-loop inside.
// Weights pre-packed to WMMA B-fragment layout in d_ws by a prep kernel.
// This revision: branch-free tanh/sigmoid on v_exp_f32/v_rcp_f32 (the libm
// tanhf path was emitting EXEC-divergent two-way code inside the hot loop),
// and explicit 128-bit vector loads for the x-fragment and obs reduction.
// ---------------------------------------------------------------------------

typedef __attribute__((ext_vector_type(16))) _Float16 v16h;
typedef __attribute__((ext_vector_type(8)))  _Float16 v8h;
typedef __attribute__((ext_vector_type(8)))  float    v8f;
typedef __attribute__((ext_vector_type(4)))  float    v4f;

#define TT  50
#define BB  32
#define VV  64
#define DIN 32
#define DR  256
#define ROWSTRIDE 264   // halfs per LDS row (256 + 8 pad, keeps 16B alignment)

// workspace offsets, in halfs (fragment-packed f16 weights)
#define OFF_W0   0          // 256x256
#define OFF_W1   65536      // 256x256
#define OFF_W2   131072     // 256x256
#define OFF_WIH  196608     // 32x768
#define OFF_WHH  221184     // 256x768
#define OFF_WO   417792     // 256x32
// total 425984 halfs = 851,968 bytes

#define LOG2E 1.44269504088896340736f

__device__ __forceinline__ v8f splat8(float x) {
  v8f r;
#pragma unroll
  for (int i = 0; i < 8; ++i) r[i] = x;
  return r;
}

__device__ __forceinline__ v8f wmma_f16(v16h a, v16h b, v8f c) {
  // (neg_a, A, neg_b, B, c_mod, C, reuse_a, reuse_b)
  return __builtin_amdgcn_wmma_f32_16x16x32_f16(false, a, false, b, (short)0, c,
                                                false, false);
}

// A-operand (16x32 f16) from an LDS row-major buffer.
// Lane L (half = L>>4, M = L&15): VGPR0..3 hold K = half*8 + 0..7,
// VGPR4..7 hold K = 16 + half*8 + 0..7  -> two contiguous 16B chunks per lane.
__device__ __forceinline__ v16h load_a_lds(const _Float16* buf, int kTile, int lane) {
  const int M = lane & 15, hf = lane >> 4;
  const _Float16* p = buf + M * ROWSTRIDE + kTile * 32 + hf * 8;
  v8h lo = *(const v8h*)p;
  v8h hi = *(const v8h*)(p + 16);
  v16h r;
#pragma unroll
  for (int i = 0; i < 8; ++i) { r[i] = lo[i]; r[i + 8] = hi[i]; }
  return r;
}

// B-operand fragment, pre-packed in global memory: frag = 32 lanes x 16 halfs.
__device__ __forceinline__ v16h load_b_frag(const _Float16* frag, int lane) {
  return *(const v16h*)(frag + lane * 16);
}

// Branch-free transcendentals: v_exp_f32 + v_rcp_f32, no EXEC divergence.
// tanh(x) = 1 - 2/(e^{2x}+1); saturates to +/-1 via exp over/underflow.
__device__ __forceinline__ float fast_tanh(float x) {
  float e = __builtin_amdgcn_exp2f(x * (2.0f * LOG2E));
  return 1.0f - 2.0f * __builtin_amdgcn_rcpf(e + 1.0f);
}
__device__ __forceinline__ float sigm(float x) {
  float e = __builtin_amdgcn_exp2f(-x * LOG2E);
  return __builtin_amdgcn_rcpf(1.0f + e);
}

// ---------------------------------------------------------------------------
// Prep: pack f32 weight W[K][N] (row-major) into WMMA B-fragment order:
//   frag f = kTile*(N/16) + nTile ; within frag: lane = ((kk>>3)&1)*16 + nn,
//   slot = (kk & 7) + (kk>>4)*8  (B 32x16: lanes 0-15 K=0..7,16..23; lanes 16-31 K=8..15,24..31).
// ---------------------------------------------------------------------------
__global__ void pack_weights(const float* __restrict__ W, _Float16* __restrict__ dst,
                             int K, int N) {
  int idx = blockIdx.x * 256 + threadIdx.x;
  if (idx >= K * N) return;
  int k = idx / N, n = idx % N;
  int kt = k >> 5, kk = k & 31;
  int nt = n >> 4, nn = n & 15;
  int lane = (((kk >> 3) & 1) << 4) | nn;
  int slot = (kk & 7) | ((kk >> 4) << 3);
  size_t frag = (size_t)kt * (N >> 4) + nt;
  dst[frag * 512 + lane * 16 + slot] = (_Float16)W[idx];
}

// ---------------------------------------------------------------------------
// Main persistent kernel: blockIdx.x owns rows [16*bx, 16*bx+16).
// ---------------------------------------------------------------------------
__global__ __launch_bounds__(256, 1)
void node_step_kernel(const float* __restrict__ values, const float* __restrict__ masks,
                      const float* __restrict__ tvec,   const float* __restrict__ h0,
                      const float* __restrict__ b_ih,   const float* __restrict__ b_hh,
                      const float* __restrict__ b0,     const float* __restrict__ b1,
                      const float* __restrict__ b2,     const float* __restrict__ bo,
                      const _Float16* __restrict__ ws,
                      float* __restrict__ outXpred,
                      float* __restrict__ outPre,
                      float* __restrict__ outPost) {
  __shared__ _Float16 hA [16 * ROWSTRIDE];  // current h, f16 A-operand copy
  __shared__ _Float16 zA [16 * ROWSTRIDE];  // MLP hidden 1
  __shared__ _Float16 z2A[16 * ROWSTRIDE];  // MLP hidden 2
  __shared__ float    obsS[16];
  __shared__ float    tS[TT];

  const _Float16* fW0  = ws + OFF_W0;
  const _Float16* fW1  = ws + OFF_W1;
  const _Float16* fW2  = ws + OFF_W2;
  const _Float16* fWih = ws + OFF_WIH;
  const _Float16* fWhh = ws + OFF_WHH;
  const _Float16* fWo  = ws + OFF_WO;

  const int lane = threadIdx.x & 31;
  const int wave = threadIdx.x >> 5;
  const int rowBase = blockIdx.x * 16;
  const int nLoc = lane & 15;       // column within an N-tile (C-layout)
  const int mOff = (lane >> 4) * 8; // row offset for acc element r: M = r + mOff

  // ---- init h = broadcast(h0); stage tvec ----
  for (int i = threadIdx.x; i < 16 * DR; i += 256) {
    int M = i >> 8, c = i & 255;
    hA[M * ROWSTRIDE + c] = (_Float16)h0[c];
  }
  if (threadIdx.x < TT) tS[threadIdx.x] = tvec[threadIdx.x];

  // fp32 master copy of h at this wave's 32 columns, in registers
  float hr[2][8];
#pragma unroll
  for (int j = 0; j < 2; ++j) {
    float hv = h0[wave * 32 + j * 16 + nLoc];
#pragma unroll
    for (int r = 0; r < 8; ++r) hr[j][r] = hv;
  }

  // ---- hoist loop-invariant per-lane bias scalars ----
  float bihc[6], bhhc[6];  // [gate*2 + j]
#pragma unroll
  for (int g = 0; g < 3; ++g)
#pragma unroll
    for (int j = 0; j < 2; ++j) {
      int col = (g * 16 + wave * 2 + j) * 16 + nLoc;
      bihc[g * 2 + j] = b_ih[col];
      bhhc[g * 2 + j] = b_hh[col];
    }
  float b0c[2], b1c[2], b2c[2];
#pragma unroll
  for (int j = 0; j < 2; ++j) {
    int col = (wave * 2 + j) * 16 + nLoc;
    b0c[j] = b0[col]; b1c[j] = b1[col]; b2c[j] = b2[col];
  }
  const float boc = bo[(wave & 1) * 16 + nLoc];  // used by waves 0..1 only
  __syncthreads();

  for (int t = 0; t < TT; ++t) {
    // ---- observation flags for this step (one wave, vectorized) ----
    if (wave == 0 && lane < 16) {
      int row = rowBase + lane, b = row >> 6, v = row & 63;
      const v4f* mp = (const v4f*)(masks + ((((size_t)b * TT) + t) * VV + v) * DIN);
      float s = 0.f;
#pragma unroll
      for (int c = 0; c < 8; ++c) {
        v4f m = mp[c];
        s += fabsf(m.x) + fabsf(m.y) + fabsf(m.z) + fabsf(m.w);
      }
      obsS[lane] = (s > 1e-4f) ? 1.0f : 0.0f;
    }

    if (t > 0) {
      const float dt = (tS[t] - tS[t - 1]) * (1.0f / 3.0f);
      // ---- 3 Euler substeps: h += dt * (W2*tanh(W1*tanh(W0*h+b0)+b1)+b2) ----
      for (int sub = 0; sub < 3; ++sub) {
        // layer 0: hA -> zA
#pragma unroll
        for (int j = 0; j < 2; ++j) {
          const int nt = wave * 2 + j;
          v8f acc = splat8(b0c[j]);
#pragma unroll 4
          for (int kt = 0; kt < 8; ++kt)
            acc = wmma_f16(load_a_lds(hA, kt, lane),
                           load_b_frag(fW0 + ((size_t)(kt * 16 + nt)) * 512, lane), acc);
#pragma unroll
          for (int r = 0; r < 8; ++r)
            zA[(r + mOff) * ROWSTRIDE + nt * 16 + nLoc] = (_Float16)fast_tanh(acc[r]);
        }
        __syncthreads();
        // layer 1: zA -> z2A
#pragma unroll
        for (int j = 0; j < 2; ++j) {
          const int nt = wave * 2 + j;
          v8f acc = splat8(b1c[j]);
#pragma unroll 4
          for (int kt = 0; kt < 8; ++kt)
            acc = wmma_f16(load_a_lds(zA, kt, lane),
                           load_b_frag(fW1 + ((size_t)(kt * 16 + nt)) * 512, lane), acc);
#pragma unroll
          for (int r = 0; r < 8; ++r)
            z2A[(r + mOff) * ROWSTRIDE + nt * 16 + nLoc] = (_Float16)fast_tanh(acc[r]);
        }
        __syncthreads();
        // layer 2 + Euler update: z2A -> hr/hA
#pragma unroll
        for (int j = 0; j < 2; ++j) {
          const int nt = wave * 2 + j;
          v8f acc = splat8(b2c[j]);
#pragma unroll 4
          for (int kt = 0; kt < 8; ++kt)
            acc = wmma_f16(load_a_lds(z2A, kt, lane),
                           load_b_frag(fW2 + ((size_t)(kt * 16 + nt)) * 512, lane), acc);
#pragma unroll
          for (int r = 0; r < 8; ++r) {
            hr[j][r] += dt * acc[r];
            hA[(r + mOff) * ROWSTRIDE + nt * 16 + nLoc] = (_Float16)hr[j][r];
          }
        }
        __syncthreads();
      }
      // ---- output head: X_pred[t-1] = h1 @ Wo + bo (2 N-tiles, waves 0..1) ----
      if (wave < 2) {
        const int nt = wave;
        v8f acc = splat8(boc);
#pragma unroll 4
        for (int kt = 0; kt < 8; ++kt)
          acc = wmma_f16(load_a_lds(hA, kt, lane),
                         load_b_frag(fWo + ((size_t)(kt * 2 + nt)) * 512, lane), acc);
#pragma unroll
        for (int r = 0; r < 8; ++r) {
          int row = rowBase + r + mOff, b = row >> 6, v = row & 63;
          outXpred[((((size_t)b * (TT - 1)) + (t - 1)) * VV + v) * DIN + nt * 16 + nLoc] = acc[r];
        }
      }
    }

    // ---- store hv_pre (== current hr) ----
#pragma unroll
    for (int j = 0; j < 2; ++j) {
      const int col = wave * 32 + j * 16 + nLoc;
#pragma unroll
      for (int r = 0; r < 8; ++r) {
        int row = rowBase + r + mOff, b = row >> 6, v = row & 63;
        outPre[((((size_t)b * TT) + t) * VV + v) * DR + col] = hr[j][r];
      }
    }

    // ---- GRU, gate-sequential to bound register pressure ----
    v16h ax;  // A fragment of x = values*masks, built via 128-bit loads (K=32)
    {
      const int M = lane & 15, hf = lane >> 4;
      int row = rowBase + M, b = row >> 6, v = row & 63;
      const size_t base = ((((size_t)b * TT) + t) * VV + v) * DIN;
      const v4f* vp = (const v4f*)(values + base);
      const v4f* mp = (const v4f*)(masks + base);
      const int q = hf * 2;  // quad index of first 8-float chunk
      v4f x0 = vp[q] * mp[q];
      v4f x1 = vp[q + 1] * mp[q + 1];
      v4f x2 = vp[q + 4] * mp[q + 4];
      v4f x3 = vp[q + 5] * mp[q + 5];
#pragma unroll
      for (int i = 0; i < 4; ++i) {
        ax[i]      = (_Float16)x0[i];
        ax[i + 4]  = (_Float16)x1[i];
        ax[i + 8]  = (_Float16)x2[i];
        ax[i + 12] = (_Float16)x3[i];
      }
    }

    float rg[2][8], zg[2][8];
    // gate 0: r = sigmoid(gi_r + gh_r)
#pragma unroll
    for (int j = 0; j < 2; ++j) {
      const int nt = wave * 2 + j;  // r-gate N-tile
      v8f a = splat8(bihc[j]);
      a = wmma_f16(ax, load_b_frag(fWih + (size_t)nt * 512, lane), a);
      v8f a2 = splat8(bhhc[j]);
#pragma unroll 4
      for (int kt = 0; kt < 8; ++kt)
        a2 = wmma_f16(load_a_lds(hA, kt, lane),
                      load_b_frag(fWhh + ((size_t)(kt * 48 + nt)) * 512, lane), a2);
#pragma unroll
      for (int r = 0; r < 8; ++r) rg[j][r] = sigm(a[r] + a2[r]);
    }
    // gate 1: z = sigmoid(gi_z + gh_z)
#pragma unroll
    for (int j = 0; j < 2; ++j) {
      const int nt = 16 + wave * 2 + j;  // z-gate N-tile
      v8f a = splat8(bihc[2 + j]);
      a = wmma_f16(ax, load_b_frag(fWih + (size_t)nt * 512, lane), a);
      v8f a2 = splat8(bhhc[2 + j]);
#pragma unroll 4
      for (int kt = 0; kt < 8; ++kt)
        a2 = wmma_f16(load_a_lds(hA, kt, lane),
                      load_b_frag(fWhh + ((size_t)(kt * 48 + nt)) * 512, lane), a2);
#pragma unroll
      for (int r = 0; r < 8; ++r) zg[j][r] = sigm(a[r] + a2[r]);
    }
    // gate 2: n = tanh(gi_n + r*gh_n)
    float ng[2][8];
#pragma unroll
    for (int j = 0; j < 2; ++j) {
      const int nt = 32 + wave * 2 + j;  // n-gate N-tile
      v8f a = splat8(bihc[4 + j]);
      a = wmma_f16(ax, load_b_frag(fWih + (size_t)nt * 512, lane), a);
      v8f a2 = splat8(bhhc[4 + j]);
#pragma unroll 4
      for (int kt = 0; kt < 8; ++kt)
        a2 = wmma_f16(load_a_lds(hA, kt, lane),
                      load_b_frag(fWhh + ((size_t)(kt * 48 + nt)) * 512, lane), a2);
#pragma unroll
      for (int r = 0; r < 8; ++r)
        ng[j][r] = fast_tanh(a[r] + rg[j][r] * a2[r]);
    }
    __syncthreads();  // all reads of hA complete

    // ---- observation blend, write-back (all in-register per wave) ----
#pragma unroll
    for (int j = 0; j < 2; ++j) {
      const int col = wave * 32 + j * 16 + nLoc;
#pragma unroll
      for (int r = 0; r < 8; ++r) {
        float h1 = hr[j][r];
        float hg = (1.0f - zg[j][r]) * ng[j][r] + zg[j][r] * h1;
        const int M = r + mOff;
        float hp = h1 + obsS[M] * (hg - h1);
        hr[j][r] = hp;
        hA[M * ROWSTRIDE + col] = (_Float16)hp;
        int row = rowBase + M, b = row >> 6, v = row & 63;
        outPost[((((size_t)b * TT) + t) * VV + v) * DR + col] = hp;
      }
    }
    __syncthreads();  // hA stable for next step
  }
}

// ---------------------------------------------------------------------------
extern "C" void kernel_launch(void* const* d_in, const int* in_sizes, int n_in,
                              void* d_out, int out_size, void* d_ws, size_t ws_size,
                              hipStream_t stream) {
  (void)in_sizes; (void)n_in; (void)out_size; (void)ws_size;
  const float* values = (const float*)d_in[0];
  const float* masks  = (const float*)d_in[1];
  const float* tvec   = (const float*)d_in[2];
  const float* h0     = (const float*)d_in[3];
  const float* W_ih   = (const float*)d_in[4];
  const float* W_hh   = (const float*)d_in[5];
  const float* b_ih   = (const float*)d_in[6];
  const float* b_hh   = (const float*)d_in[7];
  const float* W0     = (const float*)d_in[8];
  const float* b0     = (const float*)d_in[9];
  const float* W1     = (const float*)d_in[10];
  const float* b1     = (const float*)d_in[11];
  const float* W2     = (const float*)d_in[12];
  const float* b2     = (const float*)d_in[13];
  const float* Wo     = (const float*)d_in[14];
  const float* bo     = (const float*)d_in[15];

  _Float16* ws = (_Float16*)d_ws;
  auto pack = [&](const float* W, size_t off, int K, int N) {
    int n = K * N;
    pack_weights<<<(n + 255) / 256, 256, 0, stream>>>(W, ws + off, K, N);
  };
  pack(W0,   OFF_W0,  256, 256);
  pack(W1,   OFF_W1,  256, 256);
  pack(W2,   OFF_W2,  256, 256);
  pack(W_ih, OFF_WIH,  32, 768);
  pack(W_hh, OFF_WHH, 256, 768);
  pack(Wo,   OFF_WO,  256,  32);

  float* outX    = (float*)d_out;
  float* outPre  = outX   + (size_t)BB * (TT - 1) * VV * DIN;
  float* outPost = outPre + (size_t)BB * TT * VV * DR;

  node_step_kernel<<<128, 256, 0, stream>>>(values, masks, tvec, h0,
                                            b_ih, b_hh, b0, b1, b2, bo,
                                            ws, outX, outPre, outPost);
}
